// SMPNLayer_82592221102879
// MI455X (gfx1250) — matrix operand order repaired and checked by reading, not compiled, not run on hardware
//
#include <hip/hip_runtime.h>

#define N_NODES   50000
#define N_EDGES   800000
#define NODE_D    64
#define EDGE_D    32
#define ST_D      32
#define HID_D     128
#define MSG_IN_D  192   /* 2*64+32+32 */
#define UPD_IN_D  96    /* 64+32 */

#define SA_STRIDE  200  /* 192 + 8 pad (bf16 elems), 400B: 16B aligned */
#define SH_STRIDE  136  /* 128 + 8 pad, 272B: 16B aligned */
#define SU_STRIDE  104  /* 96 + 8 pad, 208B: 16B aligned */

typedef __attribute__((ext_vector_type(16))) __bf16 v16bf;
typedef __attribute__((ext_vector_type(8)))  float  v8f;

union Frag16 {
    v16bf v;
    unsigned short u[16];
};

__device__ __forceinline__ unsigned short f2bf(float f) {
    __bf16 h = (__bf16)f;                       // native v_cvt on gfx1250
    return __builtin_bit_cast(unsigned short, h);
}

// pack 4 floats -> 4 bf16, one 8B LDS store
__device__ __forceinline__ void store_bf16x4(unsigned short* dst, float4 v) {
    unsigned int lo = (unsigned int)f2bf(v.x) | ((unsigned int)f2bf(v.y) << 16);
    unsigned int hi = (unsigned int)f2bf(v.z) | ((unsigned int)f2bf(v.w) << 16);
    *(uint2*)dst = make_uint2(lo, hi);
}

// ---- fragment loaders ------------------------------------------------------
// A (16x32 bf16, MxK): lanes 0-15 row M=lane hold K={k0+0..7, k0+16..23};
// lanes 16-31 row M=lane-16 hold K={k0+8..15, k0+24..31}. Two 16B chunks/lane.
__device__ __forceinline__ v16bf load_a_frag(const unsigned short* s,
                                             int stride, int mBase, int k0, int lane) {
    const int row  = mBase + (lane & 15);
    const int half = lane >> 4;
    const unsigned short* p = s + row * stride + k0 + 8 * half;
    Frag16 f;
    *(uint4*)&f.u[0] = *(const uint4*)(p);
    *(uint4*)&f.u[8] = *(const uint4*)(p + 16);
    return f.v;
}

// B (32x16 bf16, KxN) from W^T (N x K, row-contiguous in K):
// lane holds column N=lane&15, K = {k0+8*(lane>>4)+0..7, k0+16+8*(lane>>4)+0..7}.
__device__ __forceinline__ v16bf load_b_frag(const unsigned short* wT,
                                             int K, int nBase, int k0, int lane) {
    const int col  = lane & 15;
    const int half = lane >> 4;
    const unsigned short* p = wT + (size_t)(nBase + col) * K + k0 + 8 * half;
    Frag16 f;
    *(uint4*)&f.u[0] = *(const uint4*)(p);
    *(uint4*)&f.u[8] = *(const uint4*)(p + 16);
    return f.v;
}

// One dense GEMM layer over a 64-row LDS tile: out = relu(in @ W + b), bf16 out.
// M = 64 (4 tiles of 16), 8 waves. Wave w owns column tile ni = w % Nt and
// row tiles mi = w/Nt + k*(8/Nt): B fragments are loaded ONCE per wave.
template <int K, int Nt>
__device__ __forceinline__ void mlp_layer_lds(const unsigned short* sIn, int strideIn,
                                              const unsigned short* wT,
                                              const float* bias,
                                              unsigned short* sOut, int strideOut,
                                              int wave, int lane) {
    constexpr int KF     = K / 32;
    constexpr int miStep = 8 / Nt;
    constexpr int nIter  = (4 * Nt) / 8;
    const int ni   = wave % Nt;
    const int mi0  = wave / Nt;
    const int half = lane >> 4;
    const int cl   = lane & 15;

    v16bf bf[KF];
    #pragma unroll
    for (int kf = 0; kf < KF; ++kf)
        bf[kf] = load_b_frag(wT, K, ni * 16, kf * 32, lane);

    const int   col = ni * 16 + cl;
    const float bv  = bias[col];

    #pragma unroll
    for (int it = 0; it < nIter; ++it) {
        const int mi = mi0 + it * miStep;
        v8f c = {};
        #pragma unroll
        for (int kf = 0; kf < KF; ++kf) {
            v16bf a = load_a_frag(sIn, strideIn, mi * 16, kf * 32, lane);
            c = __builtin_amdgcn_wmma_f32_16x16x32_bf16(false, a, false, bf[kf],
                                                        (short)0, c, false, false);
        }
        #pragma unroll
        for (int r = 0; r < 8; ++r) {
            const int row = mi * 16 + 8 * half + r;   // C layout: M=r (+8 hi half)
            sOut[row * strideOut + col] = f2bf(fmaxf(c[r] + bv, 0.f));
        }
    }
}

// ---- kernels ---------------------------------------------------------------
__global__ void convert_transpose(const float* __restrict__ src,
                                  unsigned short* __restrict__ dst, int K, int N) {
    int idx = blockIdx.x * blockDim.x + threadIdx.x;
    if (idx >= K * N) return;
    int n = idx / K, k = idx % K;
    dst[(size_t)n * K + k] = f2bf(src[(size_t)k * N + n]);   // dst = W^T in bf16
}

__global__ void zero_kernel(float* __restrict__ p, int n) {
    int i = blockIdx.x * blockDim.x + threadIdx.x;
    if (i < n) p[i] = 0.f;
}

__global__ __launch_bounds__(256)
void edge_msg_kernel(const float* __restrict__ nf, const float* __restrict__ ef,
                     const float* __restrict__ st, const int* __restrict__ eidx,
                     const unsigned short* __restrict__ w1t, const float* __restrict__ b1,
                     const unsigned short* __restrict__ w2t, const float* __restrict__ b2,
                     const unsigned short* __restrict__ w3t, const float* __restrict__ b3,
                     float* __restrict__ agg) {
    __shared__ unsigned short sA [64 * SA_STRIDE];
    __shared__ unsigned short sH1[64 * SH_STRIDE];
    __shared__ unsigned short sH2[64 * SH_STRIDE];
    __shared__ int            sSrc[64];
    __shared__ int            sDst[64];

    const int tid  = threadIdx.x;
    const int lane = tid & 31;
    const int wave = tid >> 5;
    const int e0   = blockIdx.x * 64;

    // Stage edge endpoints (one b64 load per edge).
    if (tid < 64) {
        int2 p = ((const int2*)eidx)[e0 + tid];
        sSrc[tid] = p.x;
        sDst[tid] = p.y;
    }
    __syncthreads();

    // Gather msg_in = [nf[src] | nf[dst] | ef | st[src]] as bf16, float4 chunks.
    // 4 threads/edge; thread owns 12 of the 48 float4 chunks (f = 4c..4c+3).
    {
        const int e    = tid >> 2;
        const int part = tid & 3;
        const int s    = sSrc[e];
        const int d    = sDst[e];
        const size_t edge = (size_t)(e0 + e);
        unsigned short* rowp = &sA[e * SA_STRIDE];
        #pragma unroll
        for (int i = 0; i < 12; ++i) {
            const int c = part * 12 + i;
            float4 v;
            if      (c < 16) v = ((const float4*)(nf + (size_t)s * NODE_D))[c];
            else if (c < 32) v = ((const float4*)(nf + (size_t)d * NODE_D))[c - 16];
            else if (c < 40) v = ((const float4*)(ef + edge * EDGE_D))[c - 32];
            else             v = ((const float4*)(st + (size_t)s * ST_D))[c - 40];
            store_bf16x4(rowp + 4 * c, v);
        }
    }
    __syncthreads();

    // Layer 1: 64x192 @ 192x128, ReLU
    mlp_layer_lds<MSG_IN_D, 8>(sA, SA_STRIDE, w1t, b1, sH1, SH_STRIDE, wave, lane);
    __syncthreads();
    // Layer 2: 64x128 @ 128x128, ReLU
    mlp_layer_lds<HID_D, 8>(sH1, SH_STRIDE, w2t, b2, sH2, SH_STRIDE, wave, lane);
    __syncthreads();

    // Layer 3: 64x128 @ 128x64, scatter-add into aggregated[dst].
    {
        const int half = lane >> 4;
        const int cl   = lane & 15;
        const int ni   = wave % 4;        // Nt = 4
        const int mi0  = wave / 4;
        v16bf bf[4];
        #pragma unroll
        for (int kf = 0; kf < 4; ++kf)
            bf[kf] = load_b_frag(w3t, HID_D, ni * 16, kf * 32, lane);
        const int   col = ni * 16 + cl;
        const float bv  = b3[col];
        #pragma unroll
        for (int it = 0; it < 2; ++it) {
            const int mi = mi0 + it * 2;
            v8f c = {};
            #pragma unroll
            for (int kf = 0; kf < 4; ++kf) {
                v16bf a = load_a_frag(sH2, SH_STRIDE, mi * 16, kf * 32, lane);
                c = __builtin_amdgcn_wmma_f32_16x16x32_bf16(false, a, false, bf[kf],
                                                            (short)0, c, false, false);
            }
            #pragma unroll
            for (int r = 0; r < 8; ++r) {
                const int row  = mi * 16 + 8 * half + r;
                const int node = sDst[row];
                atomicAdd(&agg[(size_t)node * NODE_D + col], c[r] + bv);
            }
        }
    }
}

__global__ __launch_bounds__(256)
void node_update_kernel(const float* __restrict__ nf, const float* __restrict__ st,
                        const unsigned short* __restrict__ u1t, const float* __restrict__ ub1,
                        const unsigned short* __restrict__ u2t, const float* __restrict__ ub2,
                        float* __restrict__ out) {
    __shared__ unsigned short sA[64 * SU_STRIDE];
    __shared__ unsigned short sH[64 * SH_STRIDE];

    const int tid  = threadIdx.x;
    const int lane = tid & 31;
    const int wave = tid >> 5;
    const int n0   = blockIdx.x * 64;

    // Gather upd_in = [nf | st] as bf16 (zero rows past N_NODES), float4 chunks.
    // 4 threads/node, 6 chunks each of the 24 (96 feats).
    {
        const int row  = tid >> 2;
        const int part = tid & 3;
        const int node = n0 + row;
        const bool ok  = node < N_NODES;
        unsigned short* rowp = &sA[row * SU_STRIDE];
        #pragma unroll
        for (int i = 0; i < 6; ++i) {
            const int c = part * 6 + i;
            float4 v = make_float4(0.f, 0.f, 0.f, 0.f);
            if (ok) {
                if (c < 16) v = ((const float4*)(nf + (size_t)node * NODE_D))[c];
                else        v = ((const float4*)(st + (size_t)node * ST_D))[c - 16];
            }
            store_bf16x4(rowp + 4 * c, v);
        }
    }
    __syncthreads();

    // Layer 1: 64x96 @ 96x128, ReLU
    mlp_layer_lds<UPD_IN_D, 8>(sA, SU_STRIDE, u1t, ub1, sH, SH_STRIDE, wave, lane);
    __syncthreads();

    // Layer 2: 64x128 @ 128x64, residual add, f32 store.
    {
        const int half = lane >> 4;
        const int cl   = lane & 15;
        const int ni   = wave % 4;        // Nt = 4
        const int mi0  = wave / 4;
        v16bf bf[4];
        #pragma unroll
        for (int kf = 0; kf < 4; ++kf)
            bf[kf] = load_b_frag(u2t, HID_D, ni * 16, kf * 32, lane);
        const int   col = ni * 16 + cl;
        const float bv  = ub2[col];
        #pragma unroll
        for (int it = 0; it < 2; ++it) {
            const int mi = mi0 + it * 2;
            v8f c = {};
            #pragma unroll
            for (int kf = 0; kf < 4; ++kf) {
                v16bf a = load_a_frag(sH, SH_STRIDE, mi * 16, kf * 32, lane);
                c = __builtin_amdgcn_wmma_f32_16x16x32_bf16(false, a, false, bf[kf],
                                                            (short)0, c, false, false);
            }
            #pragma unroll
            for (int r = 0; r < 8; ++r) {
                const int row  = mi * 16 + 8 * half + r;
                const int node = n0 + row;
                if (node < N_NODES)
                    out[(size_t)node * NODE_D + col] =
                        nf[(size_t)node * NODE_D + col] + c[r] + bv;
            }
        }
    }
}

// ---- host ------------------------------------------------------------------
extern "C" void kernel_launch(void* const* d_in, const int* in_sizes, int n_in,
                              void* d_out, int out_size, void* d_ws, size_t ws_size,
                              hipStream_t stream) {
    const float* nf  = (const float*)d_in[0];
    const float* ef  = (const float*)d_in[1];
    const float* st  = (const float*)d_in[2];
    const float* mW1 = (const float*)d_in[3];
    const float* mb1 = (const float*)d_in[4];
    const float* mW2 = (const float*)d_in[5];
    const float* mb2 = (const float*)d_in[6];
    const float* mW3 = (const float*)d_in[7];
    const float* mb3 = (const float*)d_in[8];
    const float* uW1 = (const float*)d_in[9];
    const float* ub1 = (const float*)d_in[10];
    const float* uW2 = (const float*)d_in[11];
    const float* ub2 = (const float*)d_in[12];
    const int*  eidx = (const int*)d_in[13];

    unsigned short* w1t = (unsigned short*)d_ws;        // 128 x 192
    unsigned short* w2t = w1t + 128 * 192;              // 128 x 128
    unsigned short* w3t = w2t + 128 * 128;              //  64 x 128
    unsigned short* u1t = w3t + 64 * 128;               // 128 x  96
    unsigned short* u2t = u1t + 128 * 96;               //  64 x 128

    float* out_nodes = (float*)d_out;
    float* agg       = out_nodes + (size_t)N_NODES * NODE_D;

    // bf16 transposed weights
    convert_transpose<<<(192 * 128 + 255) / 256, 256, 0, stream>>>(mW1, w1t, MSG_IN_D, HID_D);
    convert_transpose<<<(128 * 128 + 255) / 256, 256, 0, stream>>>(mW2, w2t, HID_D, HID_D);
    convert_transpose<<<(128 *  64 + 255) / 256, 256, 0, stream>>>(mW3, w3t, HID_D, NODE_D);
    convert_transpose<<<( 96 * 128 + 255) / 256, 256, 0, stream>>>(uW1, u1t, UPD_IN_D, HID_D);
    convert_transpose<<<(128 *  64 + 255) / 256, 256, 0, stream>>>(uW2, u2t, HID_D, NODE_D);

    // aggregated output must start at zero (atomic accumulation target)
    zero_kernel<<<(N_NODES * NODE_D + 255) / 256, 256, 0, stream>>>(agg, N_NODES * NODE_D);

    edge_msg_kernel<<<N_EDGES / 64, 256, 0, stream>>>(nf, ef, st, eidx,
                                                      w1t, mb1, w2t, mb2, w3t, mb3, agg);
    node_update_kernel<<<(N_NODES + 63) / 64, 256, 0, stream>>>(nf, st,
                                                                u1t, ub1, u2t, ub2,
                                                                out_nodes);
}
